// Net_84954453115043
// MI455X (gfx1250) — matrix operand order repaired and checked by reading, not compiled, not run on hardware
//
#include <hip/hip_runtime.h>
#include <hip/hip_bf16.h>
#include <math.h>

typedef __bf16        v16bf __attribute__((ext_vector_type(16)));
typedef float         v8f   __attribute__((ext_vector_type(8)));
typedef unsigned int  v8u   __attribute__((ext_vector_type(8)));

// ---------- helpers ----------
static __device__ __forceinline__ unsigned bf16pk(float lo, float hi) {
    // round-to-nearest-even f32 -> bf16, pack two into a dword (lo in [15:0])
    unsigned a = __float_as_uint(lo), b = __float_as_uint(hi);
    a = a + 0x7FFFu + ((a >> 16) & 1u);
    b = b + 0x7FFFu + ((b >> 16) & 1u);
    return (a >> 16) | (b & 0xFFFF0000u);
}

static __device__ __forceinline__ v8f vzero8() {
    v8f v;
    for (int i = 0; i < 8; ++i) v[i] = 0.0f;
    return v;
}

static __device__ __forceinline__ v8f wmma_bf16(v8u a, v8u b, v8f c) {
    return __builtin_amdgcn_wmma_f32_16x16x32_bf16(
        false, __builtin_bit_cast(v16bf, a),
        false, __builtin_bit_cast(v16bf, b),
        (short)0, c, false, false);
}

// Pack weights (row-major [K,N] f32) into per-lane WMMA B fragments (bf16).
// Fragment layout assumed: lane<16 -> column n = nt*16+lane, K = kt*32 + 2v+(v>=4?8:0), +1
//                          lane>=16 -> same column, K = kt*32 + 8 + 2v+(v>=4?8:0), +1
__global__ void pack_weights_kernel(const float* __restrict__ Wg1,
                                    const float* __restrict__ Wg2,
                                    const float* __restrict__ Wf1,
                                    unsigned* __restrict__ p1,
                                    unsigned* __restrict__ p2,
                                    unsigned* __restrict__ pf1) {
    int idx = blockIdx.x * blockDim.x + threadIdx.x;
    const float* W; unsigned* P; int NT, Kr, Nc, e;
    if (idx < 7168)       { e = idx;        W = Wg1; P = p1;  NT = 7; Kr = 128; Nc = 100; }
    else if (idx < 9216)  { e = idx - 7168; W = Wg2; P = p2;  NT = 2; Kr = 100; Nc = 20;  }
    else if (idx < 17408) { e = idx - 9216; W = Wf1; P = pf1; NT = 8; Kr = 128; Nc = 128; }
    else return;
    int v    = e & 7;
    int lane = (e >> 3) & 31;
    int rest = e >> 8;
    int nt   = rest % NT;
    int kt   = rest / NT;
    int n = nt * 16 + (lane & 15);
    int k = kt * 32 + ((lane & 16) ? 8 : 0) + 2 * v + ((v >= 4) ? 8 : 0);
    float lo = (k     < Kr && n < Nc) ? W[(size_t)k       * Nc + n] : 0.0f;
    float hi = (k + 1 < Kr && n < Nc) ? W[(size_t)(k + 1) * Nc + n] : 0.0f;
    P[e] = bf16pk(lo, hi);
}

// ---------- GCN layer 1 scatter: accum[dst] += feat[src]; deg[dst] += 1 ----------
__global__ void scatter_feat_kernel(const float* __restrict__ feat,
                                    const int* __restrict__ src,
                                    const int* __restrict__ dst,
                                    float* __restrict__ accum,
                                    float* __restrict__ deg, long long E) {
    long long idx = (long long)blockIdx.x * blockDim.x + threadIdx.x;
    long long e = idx >> 5;
    int lane = (int)(idx & 31);
    if (e >= E) return;
    int s = src[e], d = dst[e];
    const float4 v = *(const float4*)(feat + (size_t)s * 128 + lane * 4);
    float* a = accum + (size_t)d * 128 + lane * 4;
    atomicAdd(a + 0, v.x);
    atomicAdd(a + 1, v.y);
    atomicAdd(a + 2, v.z);
    atomicAdd(a + 3, v.w);
    if (lane == 0) atomicAdd(deg + d, 1.0f);
}

// ---------- Fused: h1 = relu((accum/deg)@Wg1 + bg1); t2 = h1@Wg2 ----------
__global__ void __launch_bounds__(128) node_gemm_kernel(
    const float* __restrict__ accum1, const float* __restrict__ deg,
    const unsigned* __restrict__ p1, const unsigned* __restrict__ p2,
    const float* __restrict__ bg1, float* __restrict__ t2, int nTiles) {
    __shared__ float hsh[4][16][128];
    const int lane = threadIdx.x & 31;
    const int wv   = threadIdx.x >> 5;
    const int tile = blockIdx.x * 4 + wv;
    if (tile >= nTiles) return;
    const int m0   = tile * 16;
    const int r    = m0 + (lane & 15);
    const int hk   = (lane & 16) ? 8 : 0;   // A-layout K sub-offset
    const int roff = (lane & 16) ? 8 : 0;   // C-layout row sub-offset
    const float invd = 1.0f / fmaxf(deg[r], 1.0f);
    const float* arow = accum1 + (size_t)r * 128;

    v8f C[7];
    #pragma unroll
    for (int i = 0; i < 7; ++i) C[i] = vzero8();

    #pragma unroll
    for (int kt = 0; kt < 4; ++kt) {
        const int kb = kt * 32 + hk;
        float4 a0 = *(const float4*)(arow + kb);
        float4 a1 = *(const float4*)(arow + kb + 4);
        float4 a2 = *(const float4*)(arow + kb + 16);
        float4 a3 = *(const float4*)(arow + kb + 20);
        v8u A;
        A[0] = bf16pk(a0.x * invd, a0.y * invd);
        A[1] = bf16pk(a0.z * invd, a0.w * invd);
        A[2] = bf16pk(a1.x * invd, a1.y * invd);
        A[3] = bf16pk(a1.z * invd, a1.w * invd);
        A[4] = bf16pk(a2.x * invd, a2.y * invd);
        A[5] = bf16pk(a2.z * invd, a2.w * invd);
        A[6] = bf16pk(a3.x * invd, a3.y * invd);
        A[7] = bf16pk(a3.z * invd, a3.w * invd);
        #pragma unroll
        for (int nt = 0; nt < 7; ++nt) {
            v8u Bf = *(const v8u*)(p1 + (((kt * 7 + nt) * 32 + lane) * 8));
            C[nt] = wmma_bf16(A, Bf, C[nt]);
        }
    }

    // bias + relu -> LDS tile (padded to 128 cols with zeros)
    #pragma unroll
    for (int nt = 0; nt < 7; ++nt) {
        int col = nt * 16 + (lane & 15);
        float bias = (col < 100) ? bg1[col] : 0.0f;
        #pragma unroll
        for (int i = 0; i < 8; ++i)
            hsh[wv][i + roff][col] = fmaxf(C[nt][i] + bias, 0.0f);
    }
    #pragma unroll
    for (int i = 0; i < 8; ++i)
        hsh[wv][i + roff][112 + (lane & 15)] = 0.0f;

    // GEMM2: t2 = h1 @ Wg2 (no bias/relu here; those apply after aggregation)
    v8f C2[2];
    C2[0] = vzero8();
    C2[1] = vzero8();
    const float* hrow = &hsh[wv][lane & 15][0];
    #pragma unroll
    for (int kt = 0; kt < 4; ++kt) {
        const int kb = kt * 32 + hk;
        float4 a0 = *(const float4*)(hrow + kb);
        float4 a1 = *(const float4*)(hrow + kb + 4);
        float4 a2 = *(const float4*)(hrow + kb + 16);
        float4 a3 = *(const float4*)(hrow + kb + 20);
        v8u A;
        A[0] = bf16pk(a0.x, a0.y); A[1] = bf16pk(a0.z, a0.w);
        A[2] = bf16pk(a1.x, a1.y); A[3] = bf16pk(a1.z, a1.w);
        A[4] = bf16pk(a2.x, a2.y); A[5] = bf16pk(a2.z, a2.w);
        A[6] = bf16pk(a3.x, a3.y); A[7] = bf16pk(a3.z, a3.w);
        #pragma unroll
        for (int nt = 0; nt < 2; ++nt) {
            v8u Bf = *(const v8u*)(p2 + (((kt * 2 + nt) * 32 + lane) * 8));
            C2[nt] = wmma_bf16(A, Bf, C2[nt]);
        }
    }
    #pragma unroll
    for (int nt = 0; nt < 2; ++nt) {
        int col = nt * 16 + (lane & 15);
        if (col < 20) {
            #pragma unroll
            for (int i = 0; i < 8; ++i)
                t2[(size_t)(m0 + i + roff) * 20 + col] = C2[nt][i];
        }
    }
}

// ---------- GCN layer 2 scatter (20 dims) ----------
__global__ void scatter_t2_kernel(const float* __restrict__ t2,
                                  const int* __restrict__ src,
                                  const int* __restrict__ dst,
                                  float* __restrict__ accum2, long long E) {
    long long idx = (long long)blockIdx.x * blockDim.x + threadIdx.x;
    long long e = idx >> 5;
    int lane = (int)(idx & 31);
    if (e >= E || lane >= 20) return;
    int s = src[e], d = dst[e];
    atomicAdd(accum2 + (size_t)d * 20 + lane, t2[(size_t)s * 20 + lane]);
}

// ---------- h2 = relu(accum2/deg + bg2); pool per graph ----------
__global__ void pool_kernel(const float* __restrict__ accum2,
                            const float* __restrict__ deg,
                            const float* __restrict__ bg2,
                            const int* __restrict__ gid,
                            float* __restrict__ hg_sum,
                            float* __restrict__ cnt, int N) {
    long long idx = (long long)blockIdx.x * blockDim.x + threadIdx.x;
    int n = (int)(idx >> 5);
    int lane = (int)(idx & 31);
    if (n >= N) return;
    int g = gid[n];
    if (lane < 20) {
        float v = fmaxf(accum2[(size_t)n * 20 + lane] / fmaxf(deg[n], 1.0f) + bg2[lane], 0.0f);
        atomicAdd(hg_sum + (size_t)g * 20 + lane, v);
    } else if (lane == 20) {
        atomicAdd(cnt + g, 1.0f);
    }
}

// ---------- gated fusion: fusion[b] = [h_g, sigmoid((h_g@W2)*h_d)*h_d] ----------
__global__ void __launch_bounds__(64) fusion_kernel(
    const float* __restrict__ hg_sum, const float* __restrict__ cnt,
    const float* __restrict__ desc2d,
    const float* __restrict__ Wpg, const float* __restrict__ bpg,
    const float* __restrict__ Wp2, const float* __restrict__ bp2,
    const float* __restrict__ W2, float* __restrict__ fusion) {
    int b = blockIdx.x, j = threadIdx.x;
    __shared__ float shg[20];
    __shared__ float shv[64];
    if (j < 20) shg[j] = hg_sum[(size_t)b * 20 + j] / fmaxf(cnt[b], 1.0f);
    __syncthreads();
    float hgj = bpg[j];
    for (int k = 0; k < 20; ++k) hgj += shg[k] * Wpg[k * 64 + j];
    float hdj = bp2[j];
    const float* drow = desc2d + (size_t)b * 200;
    for (int k = 0; k < 200; ++k) hdj += drow[k] * Wp2[k * 64 + j];
    shv[j] = hgj;
    __syncthreads();
    float t = 0.0f;
    for (int k = 0; k < 64; ++k) t += shv[k] * W2[k * 64 + j];
    float gate = 1.0f / (1.0f + __expf(-(t * hdj)));
    fusion[(size_t)b * 128 + j]      = hgj;
    fusion[(size_t)b * 128 + 64 + j] = gate * hdj;
}

// ---------- y1 = fusion @ Wf1 + bf1 (WMMA) ----------
__global__ void __launch_bounds__(256) mlp1_kernel(
    const float* __restrict__ fusion, const unsigned* __restrict__ pf1,
    const float* __restrict__ bf1, float* __restrict__ y1, int nTiles) {
    const int lane = threadIdx.x & 31;
    const int wv   = threadIdx.x >> 5;
    const int tile = blockIdx.x * 8 + wv;
    if (tile >= nTiles) return;
    const int m0   = tile * 16;
    const int r    = m0 + (lane & 15);
    const int hk   = (lane & 16) ? 8 : 0;
    const int roff = (lane & 16) ? 8 : 0;
    const float* arow = fusion + (size_t)r * 128;
    v8f C[8];
    #pragma unroll
    for (int i = 0; i < 8; ++i) C[i] = vzero8();
    #pragma unroll
    for (int kt = 0; kt < 4; ++kt) {
        const int kb = kt * 32 + hk;
        float4 a0 = *(const float4*)(arow + kb);
        float4 a1 = *(const float4*)(arow + kb + 4);
        float4 a2 = *(const float4*)(arow + kb + 16);
        float4 a3 = *(const float4*)(arow + kb + 20);
        v8u A;
        A[0] = bf16pk(a0.x, a0.y); A[1] = bf16pk(a0.z, a0.w);
        A[2] = bf16pk(a1.x, a1.y); A[3] = bf16pk(a1.z, a1.w);
        A[4] = bf16pk(a2.x, a2.y); A[5] = bf16pk(a2.z, a2.w);
        A[6] = bf16pk(a3.x, a3.y); A[7] = bf16pk(a3.z, a3.w);
        #pragma unroll
        for (int nt = 0; nt < 8; ++nt) {
            v8u Bf = *(const v8u*)(pf1 + (((kt * 8 + nt) * 32 + lane) * 8));
            C[nt] = wmma_bf16(A, Bf, C[nt]);
        }
    }
    #pragma unroll
    for (int nt = 0; nt < 8; ++nt) {
        int col = nt * 16 + (lane & 15);
        float b = bf1[col];
        #pragma unroll
        for (int i = 0; i < 8; ++i)
            y1[(size_t)(m0 + i + roff) * 128 + col] = C[nt][i] + b;
    }
}

// ---------- batchnorm statistics (training mode, biased var) ----------
__global__ void __launch_bounds__(256) bn_stats_kernel(const float* __restrict__ x,
                                                       int rows, int cols,
                                                       float* __restrict__ mu,
                                                       float* __restrict__ rs) {
    int c = blockIdx.x, t = threadIdx.x;
    __shared__ float ss[256];
    __shared__ float sq[256];
    float s = 0.0f, q = 0.0f;
    for (int r = t; r < rows; r += 256) {
        float v = x[(size_t)r * cols + c];
        s += v;
        q += v * v;
    }
    ss[t] = s; sq[t] = q;
    __syncthreads();
    for (int o = 128; o > 0; o >>= 1) {
        if (t < o) { ss[t] += ss[t + o]; sq[t] += sq[t + o]; }
        __syncthreads();
    }
    if (t == 0) {
        float m = ss[0] / rows;
        float v = sq[0] / rows - m * m;
        mu[c] = m;
        rs[c] = rsqrtf(v + 1e-5f);
    }
}

// ---------- relu(bn1(y1)) @ Wf2 + bf2 ----------
__global__ void __launch_bounds__(128) apply_mlp2_kernel(
    const float* __restrict__ y1, const float* __restrict__ mu1,
    const float* __restrict__ rs1, const float* __restrict__ g1,
    const float* __restrict__ b1, const float* __restrict__ Wf2,
    const float* __restrict__ bf2, float* __restrict__ y2) {
    int r = blockIdx.x, t = threadIdx.x;
    __shared__ float xr[128];
    float v = y1[(size_t)r * 128 + t];
    xr[t] = fmaxf(g1[t] * (v - mu1[t]) * rs1[t] + b1[t], 0.0f);
    __syncthreads();
    if (t < 32) {
        float acc = bf2[t];
        for (int k = 0; k < 128; ++k) acc += xr[k] * Wf2[k * 32 + t];
        y2[(size_t)r * 32 + t] = acc;
    }
}

// ---------- relu(bn2(y2)) @ Wf3 + bf3 ----------
__global__ void final_kernel(const float* __restrict__ y2, const float* __restrict__ mu2,
                             const float* __restrict__ rs2, const float* __restrict__ g2,
                             const float* __restrict__ b2, const float* __restrict__ Wf3,
                             const float* __restrict__ bf3, float* __restrict__ out, int B) {
    int r = blockIdx.x * blockDim.x + threadIdx.x;
    if (r >= B) return;
    float acc = bf3[0];
    for (int k = 0; k < 32; ++k) {
        float v = fmaxf(g2[k] * (y2[(size_t)r * 32 + k] - mu2[k]) * rs2[k] + b2[k], 0.0f);
        acc += v * Wf3[k];
    }
    out[r] = acc;
}

extern "C" void kernel_launch(void* const* d_in, const int* in_sizes, int n_in,
                              void* d_out, int out_size, void* d_ws, size_t ws_size,
                              hipStream_t stream) {
    const float* feat  = (const float*)d_in[0];
    const int*   src   = (const int*)d_in[1];
    const int*   dst   = (const int*)d_in[2];
    const int*   gid   = (const int*)d_in[3];
    const float* desc2 = (const float*)d_in[4];
    // d_in[5] = desc_3d (unused by reference)
    const float* Wg1 = (const float*)d_in[6];
    const float* bg1 = (const float*)d_in[7];
    const float* Wg2 = (const float*)d_in[8];
    const float* bg2 = (const float*)d_in[9];
    const float* Wpg = (const float*)d_in[10];
    const float* bpg = (const float*)d_in[11];
    const float* Wp2 = (const float*)d_in[12];
    const float* bp2 = (const float*)d_in[13];
    const float* W2  = (const float*)d_in[14];
    const float* Wf1 = (const float*)d_in[15];
    const float* bf1 = (const float*)d_in[16];
    const float* Wf2 = (const float*)d_in[17];
    const float* bf2 = (const float*)d_in[18];
    const float* Wf3 = (const float*)d_in[19];
    const float* bf3 = (const float*)d_in[20];
    const float* g1  = (const float*)d_in[21];
    const float* b1  = (const float*)d_in[22];
    const float* g2  = (const float*)d_in[23];
    const float* b2  = (const float*)d_in[24];

    const int       N = in_sizes[0] / 128;
    const long long E = in_sizes[1];
    const int       B = in_sizes[4] / 200;
    float* out = (float*)d_out;

    float* ws = (float*)d_ws;
    size_t off = 0;
    float* accum1 = ws + off;  off += (size_t)N * 128;
    float* deg    = ws + off;  off += (size_t)N;
    float* t2     = ws + off;  off += (size_t)N * 20;
    float* hg_sum = ws + off;  off += (size_t)B * 20;
    float* cnt    = ws + off;  off += (size_t)B;
    float* fusion = ws + off;  off += (size_t)B * 128;
    float* y1     = ws + off;  off += (size_t)B * 128;
    float* y2     = ws + off;  off += (size_t)B * 32;
    float* mu1    = ws + off;  off += 128;
    float* rs1    = ws + off;  off += 128;
    float* mu2    = ws + off;  off += 32;
    float* rs2    = ws + off;  off += 32;
    unsigned* p1  = (unsigned*)(ws + off); off += 7168;
    unsigned* p2  = (unsigned*)(ws + off); off += 2048;
    unsigned* pf1 = (unsigned*)(ws + off); off += 8192;
    float* accum2 = accum1;  // layer-2 accumulator reuses layer-1 space (20 <= 128 cols)

    hipMemsetAsync(accum1, 0, (size_t)N * 128 * sizeof(float), stream);
    hipMemsetAsync(deg, 0, (size_t)N * sizeof(float), stream);
    hipMemsetAsync(hg_sum, 0, ((size_t)B * 20 + B) * sizeof(float), stream);  // hg_sum + cnt

    pack_weights_kernel<<<68, 256, 0, stream>>>(Wg1, Wg2, Wf1, p1, p2, pf1);

    {
        long long tot = E * 32;
        scatter_feat_kernel<<<(unsigned)((tot + 255) / 256), 256, 0, stream>>>(
            feat, src, dst, accum1, deg, E);
    }

    int nTiles = (N + 15) / 16;
    node_gemm_kernel<<<(nTiles + 3) / 4, 128, 0, stream>>>(accum1, deg, p1, p2, bg1, t2, nTiles);

    hipMemsetAsync(accum2, 0, (size_t)N * 20 * sizeof(float), stream);
    {
        long long tot = E * 32;
        scatter_t2_kernel<<<(unsigned)((tot + 255) / 256), 256, 0, stream>>>(
            t2, src, dst, accum2, E);
    }
    {
        long long tot = (long long)N * 32;
        pool_kernel<<<(unsigned)((tot + 255) / 256), 256, 0, stream>>>(
            accum2, deg, bg2, gid, hg_sum, cnt, N);
    }

    fusion_kernel<<<B, 64, 0, stream>>>(hg_sum, cnt, desc2, Wpg, bpg, Wp2, bp2, W2, fusion);

    int mTiles = B / 16;
    mlp1_kernel<<<(mTiles + 7) / 8, 256, 0, stream>>>(fusion, pf1, bf1, y1, mTiles);

    bn_stats_kernel<<<128, 256, 0, stream>>>(y1, B, 128, mu1, rs1);
    apply_mlp2_kernel<<<B, 128, 0, stream>>>(y1, mu1, rs1, g1, b1, Wf2, bf2, y2);
    bn_stats_kernel<<<32, 256, 0, stream>>>(y2, B, 32, mu2, rs2);
    final_kernel<<<(B + 255) / 256, 256, 0, stream>>>(y2, mu2, rs2, g2, b2, Wf3, bf3, out, B);
}
